// KDABlock_1735166787622
// MI455X (gfx1250) — matrix operand order repaired and checked by reading, not compiled, not run on hardware
//
#include <hip/hip_runtime.h>
#include <hip/hip_bf16.h>
#include <math.h>

// ---------------------------------------------------------------------------
// Types for CDNA5 WMMA (wave32)
// ---------------------------------------------------------------------------
typedef __bf16 bf16_t;
typedef __attribute__((ext_vector_type(16))) __bf16 v16bf;
typedef __attribute__((ext_vector_type(8)))  float  v8f;

union Frag { v16bf v; uint4 u[2]; };

#define EPSF 1e-6f

static constexpr int MROWS = 8192;   // B*T
static constexpr int DDIM  = 1024;
static constexpr int FDIM  = 4096;
static constexpr int HHEAD = 16;
static constexpr int TSEQ  = 2048;
static constexpr int BBAT  = 4;

// ---------------------------------------------------------------------------
// Generic bf16 WMMA GEMM:  C[M,N] = epilogue(A[M,K] @ WT[N,K]^T + bias)
// block = 256 threads (8 waves). Per-wave tile: 32(M) x 64(N) = 8 WMMA tiles.
// Block tile: 256(M) x 64(N).  K must be a multiple of 64.
// Ping-pong double buffering: two fragment buffers alternate roles, so no
// register moves are needed and loads for one buffer overlap WMMAs on the
// other (staggered s_wait_loadcnt, no WAR hazard nops).
// mode: 0=none, 1=sigmoid, 2=silu
// ---------------------------------------------------------------------------
__global__ __launch_bounds__(256)
void gemm_bf16_wmma(const bf16_t* __restrict__ A, const bf16_t* __restrict__ WT,
                    float* __restrict__ C, const float* __restrict__ bias,
                    int M, int N, int K, int mode)
{
    const int lane = threadIdx.x & 31;
    const int wave = threadIdx.x >> 5;
    const int hlf  = lane >> 4;       // 0 or 1 (wave half)
    const int l16  = lane & 15;
    const int mrow = blockIdx.x * 256 + wave * 32;
    const int nb0  = blockIdx.y * 64;

    const v8f zero = {0.f,0.f,0.f,0.f,0.f,0.f,0.f,0.f};
    v8f acc[8] = {zero, zero, zero, zero, zero, zero, zero, zero};

    // A fragment bases (ISA 16x32 bf16 layout):
    // lane<16 -> row M=mrow+l16, K chunks {0..7,16..23}; lane>=16 -> {8..15,24..31}
    const bf16_t* aB[2];
    aB[0] = A + (size_t)(mrow + l16) * K + hlf * 8;
    aB[1] = aB[0] + (size_t)16 * K;

    // B fragment bases: WT[N,K] row -> lane holds N=nb+l16, K chunks same as A
    const bf16_t* bB[4];
#pragma unroll
    for (int j = 0; j < 4; ++j) {
        int n = nb0 + j * 16 + l16;
        if (n >= N) n = N - 1;        // clamp; store is guarded
        bB[j] = WT + (size_t)n * K + hlf * 8;
    }

    Frag a0[2], b0[4], a1[2], b1[4];

    auto loadF = [&](Frag* fa, Frag* fb, int kk) {
#pragma unroll
        for (int t = 0; t < 2; ++t) {
            fa[t].u[0] = *(const uint4*)(aB[t] + kk);
            fa[t].u[1] = *(const uint4*)(aB[t] + kk + 16);
        }
#pragma unroll
        for (int j = 0; j < 4; ++j) {
            fb[j].u[0] = *(const uint4*)(bB[j] + kk);
            fb[j].u[1] = *(const uint4*)(bB[j] + kk + 16);
        }
    };
    auto mmaF = [&](const Frag* fa, const Frag* fb) {
#pragma unroll
        for (int t = 0; t < 2; ++t)
#pragma unroll
            for (int j = 0; j < 4; ++j)
                acc[t * 4 + j] = __builtin_amdgcn_wmma_f32_16x16x32_bf16(
                    false, fa[t].v, false, fb[j].v, (short)0, acc[t * 4 + j],
                    false, false);
    };

    loadF(a0, b0, 0);
    for (int k0 = 0; k0 < K; k0 += 64) {
        loadF(a1, b1, k0 + 32);           // always valid: K % 64 == 0
        mmaF(a0, b0);
        if (k0 + 64 < K) loadF(a0, b0, k0 + 64);
        mmaF(a1, b1);
    }

#pragma unroll
    for (int t = 0; t < 2; ++t) {
        const int cmBase = mrow + t * 16 + hlf * 8;
#pragma unroll
        for (int j = 0; j < 4; ++j) {
            int cn = nb0 + j * 16 + l16;
            if (cn >= N) continue;
            float bso = (bias != nullptr) ? bias[cn] : 0.f;
#pragma unroll
            for (int i = 0; i < 8; ++i) {
                float s = acc[t * 4 + j][i] + bso;
                if (mode == 1)      s = 1.f / (1.f + __expf(-s));
                else if (mode == 2) s = s / (1.f + __expf(-s));
                C[(size_t)(cmBase + i) * N + cn] = s;
            }
        }
    }
}

// ---------------------------------------------------------------------------
// Fused FFN up-projection: GB = bf16( silu(A@W1T^T) * (A@W3T^T) )
// Per-wave tile: 32(M) x 32(N), two weight matrices -> 8 WMMAs per k-step.
// Same ping-pong double buffering; K must be a multiple of 64.
// ---------------------------------------------------------------------------
__global__ __launch_bounds__(256)
void gemm_dual_swiglu(const bf16_t* __restrict__ A, const bf16_t* __restrict__ W1T,
                      const bf16_t* __restrict__ W3T, bf16_t* __restrict__ GB,
                      int M, int N, int K)
{
    const int lane = threadIdx.x & 31;
    const int wave = threadIdx.x >> 5;
    const int hlf  = lane >> 4;
    const int l16  = lane & 15;
    const int mrow = blockIdx.x * 256 + wave * 32;
    const int nb0  = blockIdx.y * 32;

    const v8f zero = {0.f,0.f,0.f,0.f,0.f,0.f,0.f,0.f};
    v8f acc1[4] = {zero, zero, zero, zero};   // [m tile][n tile]
    v8f acc3[4] = {zero, zero, zero, zero};

    const bf16_t* aB[2];
    aB[0] = A + (size_t)(mrow + l16) * K + hlf * 8;
    aB[1] = aB[0] + (size_t)16 * K;

    const bf16_t* b1B[2];
    const bf16_t* b3B[2];
#pragma unroll
    for (int j = 0; j < 2; ++j) {
        int n = nb0 + j * 16 + l16;
        if (n >= N) n = N - 1;
        b1B[j] = W1T + (size_t)n * K + hlf * 8;
        b3B[j] = W3T + (size_t)n * K + hlf * 8;
    }

    Frag a0[2], p0[2], q0[2], a1[2], p1[2], q1[2];

    auto loadF = [&](Frag* fa, Frag* f1, Frag* f3, int kk) {
#pragma unroll
        for (int t = 0; t < 2; ++t) {
            fa[t].u[0] = *(const uint4*)(aB[t] + kk);
            fa[t].u[1] = *(const uint4*)(aB[t] + kk + 16);
        }
#pragma unroll
        for (int j = 0; j < 2; ++j) {
            f1[j].u[0] = *(const uint4*)(b1B[j] + kk);
            f1[j].u[1] = *(const uint4*)(b1B[j] + kk + 16);
            f3[j].u[0] = *(const uint4*)(b3B[j] + kk);
            f3[j].u[1] = *(const uint4*)(b3B[j] + kk + 16);
        }
    };
    auto mmaF = [&](const Frag* fa, const Frag* f1, const Frag* f3) {
#pragma unroll
        for (int t = 0; t < 2; ++t)
#pragma unroll
            for (int j = 0; j < 2; ++j) {
                acc1[t * 2 + j] = __builtin_amdgcn_wmma_f32_16x16x32_bf16(
                    false, fa[t].v, false, f1[j].v, (short)0, acc1[t * 2 + j],
                    false, false);
                acc3[t * 2 + j] = __builtin_amdgcn_wmma_f32_16x16x32_bf16(
                    false, fa[t].v, false, f3[j].v, (short)0, acc3[t * 2 + j],
                    false, false);
            }
    };

    loadF(a0, p0, q0, 0);
    for (int k0 = 0; k0 < K; k0 += 64) {
        loadF(a1, p1, q1, k0 + 32);
        mmaF(a0, p0, q0);
        if (k0 + 64 < K) loadF(a0, p0, q0, k0 + 64);
        mmaF(a1, p1, q1);
    }

#pragma unroll
    for (int t = 0; t < 2; ++t) {
        const int cmBase = mrow + t * 16 + hlf * 8;
#pragma unroll
        for (int j = 0; j < 2; ++j) {
            int cn = nb0 + j * 16 + l16;
            if (cn >= N) continue;
#pragma unroll
            for (int i = 0; i < 8; ++i) {
                float a = acc1[t * 2 + j][i];
                float g = (a / (1.f + __expf(-a))) * acc3[t * 2 + j][i];
                GB[(size_t)(cmBase + i) * N + cn] = (bf16_t)g;
            }
        }
    }
}

// ---------------------------------------------------------------------------
// Weight prep: fp32 W[K,N] row-major -> bf16 WT[N,K]
// ---------------------------------------------------------------------------
__global__ void transpose_bf16(const float* __restrict__ W, bf16_t* __restrict__ WT,
                               int K, int N)
{
    size_t idx = (size_t)blockIdx.x * 256 + threadIdx.x;
    if (idx >= (size_t)K * N) return;
    int kk = (int)(idx / N), nn = (int)(idx % N);
    WT[(size_t)nn * K + kk] = (bf16_t)W[idx];
}

__global__ void to_bf16_kernel(const float* __restrict__ X, bf16_t* __restrict__ Y,
                               size_t n)
{
    size_t idx = (size_t)blockIdx.x * 256 + threadIdx.x;
    if (idx < n) Y[idx] = (bf16_t)X[idx];
}

// ---------------------------------------------------------------------------
// RMSNorm (one token per block, D=1024)
// ---------------------------------------------------------------------------
__global__ __launch_bounds__(256)
void rmsnorm_f32(const float* __restrict__ x, const float* __restrict__ w,
                 float* __restrict__ out, int D)
{
    int row = blockIdx.x, tid = threadIdx.x;
    const float* xr = x + (size_t)row * D;
    float s = 0.f;
    for (int i = tid; i < D; i += 256) { float v = xr[i]; s += v * v; }
    __shared__ float red[256];
    red[tid] = s; __syncthreads();
    for (int o = 128; o > 0; o >>= 1) { if (tid < o) red[tid] += red[tid + o]; __syncthreads(); }
    float rs = rsqrtf(red[0] / D + EPSF);
    for (int i = tid; i < D; i += 256) out[(size_t)row * D + i] = xr[i] * rs * w[i];
}

__global__ __launch_bounds__(256)
void rmsnorm_to_bf16(const float* __restrict__ x, const float* __restrict__ w,
                     bf16_t* __restrict__ out, int D)
{
    int row = blockIdx.x, tid = threadIdx.x;
    const float* xr = x + (size_t)row * D;
    float s = 0.f;
    for (int i = tid; i < D; i += 256) { float v = xr[i]; s += v * v; }
    __shared__ float red[256];
    red[tid] = s; __syncthreads();
    for (int o = 128; o > 0; o >>= 1) { if (tid < o) red[tid] += red[tid + o]; __syncthreads(); }
    float rs = rsqrtf(red[0] / D + EPSF);
    for (int i = tid; i < D; i += 256) out[(size_t)row * D + i] = (bf16_t)(xr[i] * rs * w[i]);
}

// ---------------------------------------------------------------------------
// Causal depthwise conv (K=4, left pad 3) + SiLU -> bf16 activations
// ---------------------------------------------------------------------------
__global__ void conv_silu_bf16(const float* __restrict__ hn, const float* __restrict__ cw,
                               const float* __restrict__ cb, bf16_t* __restrict__ hb,
                               int Bn, int T, int D)
{
    size_t idx = (size_t)blockIdx.x * 256 + threadIdx.x;
    if (idx >= (size_t)Bn * T * D) return;
    int d = (int)(idx % D);
    int t = (int)((idx / D) % T);
    int b = (int)(idx / ((size_t)D * T));
    float s = cb[d];
#pragma unroll
    for (int j = 0; j < 4; ++j) {
        int tt = t - 3 + j;
        if (tt >= 0) s += hn[((size_t)b * T + tt) * D + d] * cw[d * 4 + j];
    }
    s = s / (1.f + __expf(-s));
    hb[idx] = (bf16_t)s;
}

// ---------------------------------------------------------------------------
// Per-head L2-normalize q and k (one token per block; 16 threads/head)
// ---------------------------------------------------------------------------
__global__ __launch_bounds__(256)
void l2norm_qk(float* __restrict__ q, float* __restrict__ k)
{
    int row = blockIdx.x, tid = threadIdx.x;
    int hh = tid >> 4, l = tid & 15;
    size_t base = (size_t)row * 1024 + hh * 64 + l * 4;

    float4 qv = *(const float4*)(q + base);
    float ss = qv.x * qv.x + qv.y * qv.y + qv.z * qv.z + qv.w * qv.w;
    ss += __shfl_xor(ss, 1, 16); ss += __shfl_xor(ss, 2, 16);
    ss += __shfl_xor(ss, 4, 16); ss += __shfl_xor(ss, 8, 16);
    float inv = 1.f / (sqrtf(ss) + EPSF);
    qv.x *= inv; qv.y *= inv; qv.z *= inv; qv.w *= inv;
    *(float4*)(q + base) = qv;

    float4 kv = *(const float4*)(k + base);
    ss = kv.x * kv.x + kv.y * kv.y + kv.z * kv.z + kv.w * kv.w;
    ss += __shfl_xor(ss, 1, 16); ss += __shfl_xor(ss, 2, 16);
    ss += __shfl_xor(ss, 4, 16); ss += __shfl_xor(ss, 8, 16);
    inv = 1.f / (sqrtf(ss) + EPSF);
    kv.x *= inv; kv.y *= inv; kv.z *= inv; kv.w *= inv;
    *(float4*)(k + base) = kv;
}

// ---------------------------------------------------------------------------
// Delta-rule scan. One block per (b,h); 256 threads; thread owns S[d0:d0+16, col]
// in registers.  fp32 for numeric fidelity of the recurrence.
// ---------------------------------------------------------------------------
__global__ __launch_bounds__(256)
void delta_scan(const float* __restrict__ q, const float* __restrict__ k,
                const float* __restrict__ v, const float* __restrict__ al,
                const float* __restrict__ be, float* __restrict__ y,
                float* __restrict__ sfin, int T, int H)
{
    const int bh  = blockIdx.x;
    const int b   = bh / H, h = bh % H;
    const int tid = threadIdx.x;
    const int col = tid & 63;
    const int r   = tid >> 6;
    const int d0  = r * 16;

    float s[16];
#pragma unroll
    for (int i = 0; i < 16; ++i) s[i] = 0.f;

    __shared__ float lq[64], lk[64], lv[64], la[64];
    __shared__ float pu[4][64], po[4][64];
    __shared__ float lbeta;

    for (int t = 0; t < T; ++t) {
        size_t row = ((size_t)b * T + t) * 1024 + (size_t)h * 64;
        if (tid < 64)        lq[tid]        = q[row + tid];
        else if (tid < 128)  lk[tid - 64]   = k[row + tid - 64];
        else if (tid < 192)  lv[tid - 128]  = v[row + tid - 128];
        else                 la[tid - 192]  = al[row + tid - 192];
        if (tid == 0) lbeta = be[((size_t)b * T + t) * H + h];
        __syncthreads();

        float p = 0.f;
#pragma unroll
        for (int i = 0; i < 16; ++i) { s[i] *= la[d0 + i]; p += lk[d0 + i] * s[i]; }
        pu[r][col] = p;
        __syncthreads();

        float u = pu[0][col] + pu[1][col] + pu[2][col] + pu[3][col];
        float delta = lbeta * (lv[col] - u);
        float o = 0.f;
#pragma unroll
        for (int i = 0; i < 16; ++i) { s[i] += lk[d0 + i] * delta; o += lq[d0 + i] * s[i]; }
        po[r][col] = o;
        __syncthreads();

        if (r == 0) y[row + col] = po[0][col] + po[1][col] + po[2][col] + po[3][col];
    }
#pragma unroll
    for (int i = 0; i < 16; ++i)
        sfin[((size_t)bh * 64 + d0 + i) * 64 + col] = s[i];
}

// ---------------------------------------------------------------------------
// Headwise RMSNorm over DV=64 with per-head weight; write bf16
// ---------------------------------------------------------------------------
__global__ __launch_bounds__(256)
void head_rmsnorm_bf16(const float* __restrict__ y, const float* __restrict__ hw,
                       bf16_t* __restrict__ yb)
{
    int row = blockIdx.x, tid = threadIdx.x;
    int hh = tid >> 4, l = tid & 15;
    size_t base = (size_t)row * 1024 + hh * 64 + l * 4;
    float4 v = *(const float4*)(y + base);
    float ss = v.x * v.x + v.y * v.y + v.z * v.z + v.w * v.w;
    ss += __shfl_xor(ss, 1, 16); ss += __shfl_xor(ss, 2, 16);
    ss += __shfl_xor(ss, 4, 16); ss += __shfl_xor(ss, 8, 16);
    float rs = rsqrtf(ss / 64.f + EPSF);
    const float* w = hw + hh * 64 + l * 4;
    yb[base + 0] = (bf16_t)(v.x * rs * w[0]);
    yb[base + 1] = (bf16_t)(v.y * rs * w[1]);
    yb[base + 2] = (bf16_t)(v.z * rs * w[2]);
    yb[base + 3] = (bf16_t)(v.w * rs * w[3]);
}

// ---------------------------------------------------------------------------
// Elementwise: y1 = x + P*gate ;  out = y1 + O
// ---------------------------------------------------------------------------
__global__ void combine_gate(const float* __restrict__ x, const float* __restrict__ P,
                             const float* __restrict__ g, float* __restrict__ y1, size_t n)
{
    size_t i = (size_t)blockIdx.x * 256 + threadIdx.x;
    if (i < n) y1[i] = x[i] + P[i] * g[i];
}

__global__ void final_add(const float* __restrict__ y1, const float* __restrict__ O,
                          float* __restrict__ out, size_t n)
{
    size_t i = (size_t)blockIdx.x * 256 + threadIdx.x;
    if (i < n) out[i] = y1[i] + O[i];
}

// ---------------------------------------------------------------------------
// Workspace layout (bytes).  bf16 weights (34 MB) fit in MI455X's 192 MB L2.
// ---------------------------------------------------------------------------
static constexpr size_t MB1 = 1ull << 20;
static constexpr size_t OFF_QT  = 0;          // bf16 1024x1024  (2MB)
static constexpr size_t OFF_KT  = 2  * MB1;
static constexpr size_t OFF_VT  = 4  * MB1;
static constexpr size_t OFF_AT  = 6  * MB1;
static constexpr size_t OFF_OT  = 8  * MB1;   // out_w^T
static constexpr size_t OFF_U1T = 10 * MB1;   // [64,1024]
static constexpr size_t OFF_U2T = 11 * MB1;   // [1024,64]
static constexpr size_t OFF_BW  = 12 * MB1;   // beta_w^T [16,1024]
static constexpr size_t OFF_W1T = 13 * MB1;   // [4096,1024] 8MB
static constexpr size_t OFF_W3T = 21 * MB1;
static constexpr size_t OFF_W2T = 29 * MB1;   // [1024,4096] 8MB
static constexpr size_t OFF_HN  = 40 * MB1;   // fp32 32MB ; reused: gate
static constexpr size_t OFF_HB  = 72 * MB1;   // bf16 16MB ; reused: yb
static constexpr size_t OFF_XB  = 88 * MB1;   // bf16 16MB
static constexpr size_t OFF_Q   = 104 * MB1;  // fp32 32MB ; reused: P, then O
static constexpr size_t OFF_K   = 136 * MB1;  // fp32 32MB ; reused: y1
static constexpr size_t OFF_V   = 168 * MB1;  // fp32 32MB ; reused (+OFF_A): gb bf16 64MB
static constexpr size_t OFF_A   = 200 * MB1;  // fp32 32MB
static constexpr size_t OFF_BE  = 232 * MB1;  // fp32 512KB
static constexpr size_t OFF_Y   = 233 * MB1;  // fp32 32MB
static constexpr size_t OFF_G1  = 265 * MB1;  // fp32 2MB
static constexpr size_t OFF_G1B = 267 * MB1;  // bf16 1MB
static constexpr size_t OFF_Y2B = 268 * MB1;  // bf16 16MB  -> 284MB total

static inline int cdiv_i(size_t a, int b) { return (int)((a + b - 1) / b); }

extern "C" void kernel_launch(void* const* d_in, const int* in_sizes, int n_in,
                              void* d_out, int out_size, void* d_ws, size_t ws_size,
                              hipStream_t stream)
{
    (void)in_sizes; (void)n_in; (void)out_size; (void)ws_size;

    const float* x        = (const float*)d_in[0];
    const float* norm_in  = (const float*)d_in[1];
    const float* conv_w   = (const float*)d_in[2];
    const float* conv_b   = (const float*)d_in[3];
    const float* q_w      = (const float*)d_in[4];
    const float* k_w      = (const float*)d_in[5];
    const float* v_w      = (const float*)d_in[6];
    const float* out_w    = (const float*)d_in[7];
    const float* alpha_w  = (const float*)d_in[8];
    const float* alpha_b  = (const float*)d_in[9];
    const float* beta_w   = (const float*)d_in[10];
    const float* beta_b   = (const float*)d_in[11];
    const float* head_nw  = (const float*)d_in[12];
    const float* u1_w     = (const float*)d_in[13];
    const float* u1_b     = (const float*)d_in[14];
    const float* u2_w     = (const float*)d_in[15];
    const float* u2_b     = (const float*)d_in[16];
    const float* ff_nw    = (const float*)d_in[17];
    const float* ff_w1    = (const float*)d_in[18];
    const float* ff_w3    = (const float*)d_in[19];
    const float* ff_w2    = (const float*)d_in[20];

    float* out = (float*)d_out;
    char*  ws  = (char*)d_ws;
    auto F32 = [&](size_t off) { return (float*)(ws + off); };
    auto BF  = [&](size_t off) { return (bf16_t*)(ws + off); };

    const int M = MROWS, D = DDIM, F = FDIM, H = HHEAD, T = TSEQ, Bn = BBAT;
    const size_t MD = (size_t)M * D;
    dim3 blk(256);

    // ---- Weight prep: fp32 [K,N] -> bf16 [N,K] --------------------------------
    transpose_bf16<<<cdiv_i((size_t)D * D, 256), blk, 0, stream>>>(q_w,     BF(OFF_QT),  D, D);
    transpose_bf16<<<cdiv_i((size_t)D * D, 256), blk, 0, stream>>>(k_w,     BF(OFF_KT),  D, D);
    transpose_bf16<<<cdiv_i((size_t)D * D, 256), blk, 0, stream>>>(v_w,     BF(OFF_VT),  D, D);
    transpose_bf16<<<cdiv_i((size_t)D * D, 256), blk, 0, stream>>>(alpha_w, BF(OFF_AT),  D, D);
    transpose_bf16<<<cdiv_i((size_t)D * H, 256), blk, 0, stream>>>(beta_w,  BF(OFF_BW),  D, H);
    transpose_bf16<<<cdiv_i((size_t)D * D, 256), blk, 0, stream>>>(out_w,   BF(OFF_OT),  D, D);
    transpose_bf16<<<cdiv_i((size_t)D * 64, 256), blk, 0, stream>>>(u1_w,   BF(OFF_U1T), D, 64);
    transpose_bf16<<<cdiv_i((size_t)64 * D, 256), blk, 0, stream>>>(u2_w,   BF(OFF_U2T), 64, D);
    transpose_bf16<<<cdiv_i((size_t)D * F, 256), blk, 0, stream>>>(ff_w1,   BF(OFF_W1T), D, F);
    transpose_bf16<<<cdiv_i((size_t)D * F, 256), blk, 0, stream>>>(ff_w3,   BF(OFF_W3T), D, F);
    transpose_bf16<<<cdiv_i((size_t)F * D, 256), blk, 0, stream>>>(ff_w2,   BF(OFF_W2T), F, D);

    // ---- Input path -----------------------------------------------------------
    to_bf16_kernel<<<cdiv_i(MD, 256), blk, 0, stream>>>(x, BF(OFF_XB), MD);
    rmsnorm_f32<<<M, blk, 0, stream>>>(x, norm_in, F32(OFF_HN), D);
    conv_silu_bf16<<<cdiv_i(MD, 256), blk, 0, stream>>>(F32(OFF_HN), conv_w, conv_b,
                                                        BF(OFF_HB), Bn, T, D);

    // ---- Projections (bf16 WMMA, fp32 out) ------------------------------------
    dim3 gNN(M / 256, D / 64);       // 32 x 16 blocks
    gemm_bf16_wmma<<<gNN, blk, 0, stream>>>(BF(OFF_HB), BF(OFF_QT), F32(OFF_Q), nullptr, M, D, D, 0);
    gemm_bf16_wmma<<<gNN, blk, 0, stream>>>(BF(OFF_HB), BF(OFF_KT), F32(OFF_K), nullptr, M, D, D, 0);
    gemm_bf16_wmma<<<gNN, blk, 0, stream>>>(BF(OFF_HB), BF(OFF_VT), F32(OFF_V), nullptr, M, D, D, 0);
    gemm_bf16_wmma<<<gNN, blk, 0, stream>>>(BF(OFF_HB), BF(OFF_AT), F32(OFF_A), alpha_b, M, D, D, 1);
    gemm_bf16_wmma<<<dim3(M / 256, 1), blk, 0, stream>>>(BF(OFF_HB), BF(OFF_BW), F32(OFF_BE),
                                                         beta_b, M, H, D, 1);

    l2norm_qk<<<M, blk, 0, stream>>>(F32(OFF_Q), F32(OFF_K));

    // ---- Delta-rule recurrent scan (fp32) -------------------------------------
    float* sfin = out + MD;  // s_final directly in the output tail
    delta_scan<<<Bn * H, blk, 0, stream>>>(F32(OFF_Q), F32(OFF_K), F32(OFF_V), F32(OFF_A),
                                           F32(OFF_BE), F32(OFF_Y), sfin, T, H);

    // ---- Head norm + gated output projection ----------------------------------
    head_rmsnorm_bf16<<<M, blk, 0, stream>>>(F32(OFF_Y), head_nw, BF(OFF_HB));   // yb reuses hb
    gemm_bf16_wmma<<<dim3(M / 256, 1), blk, 0, stream>>>(BF(OFF_XB), BF(OFF_U1T), F32(OFF_G1),
                                                         u1_b, M, 64, D, 2);      // silu
    to_bf16_kernel<<<cdiv_i((size_t)M * 64, 256), blk, 0, stream>>>(F32(OFF_G1), BF(OFF_G1B),
                                                                    (size_t)M * 64);
    gemm_bf16_wmma<<<gNN, blk, 0, stream>>>(BF(OFF_G1B), BF(OFF_U2T), F32(OFF_HN),
                                            u2_b, M, D, 64, 1);                   // gate (sigmoid)
    gemm_bf16_wmma<<<gNN, blk, 0, stream>>>(BF(OFF_HB), BF(OFF_OT), F32(OFF_Q),   // P reuses q
                                            nullptr, M, D, D, 0);
    combine_gate<<<cdiv_i(MD, 256), blk, 0, stream>>>(x, F32(OFF_Q), F32(OFF_HN),
                                                      F32(OFF_K), MD);            // y1 reuses k

    // ---- FFN ------------------------------------------------------------------
    rmsnorm_to_bf16<<<M, blk, 0, stream>>>(F32(OFF_K), ff_nw, BF(OFF_Y2B), D);
    gemm_dual_swiglu<<<dim3(M / 256, F / 32), blk, 0, stream>>>(BF(OFF_Y2B), BF(OFF_W1T),
                                                                BF(OFF_W3T), BF(OFF_V), M, F, D);
    gemm_bf16_wmma<<<gNN, blk, 0, stream>>>(BF(OFF_V), BF(OFF_W2T), F32(OFF_Q),   // O reuses q
                                            nullptr, M, D, F, 0);
    final_add<<<cdiv_i(MD, 256), blk, 0, stream>>>(F32(OFF_K), F32(OFF_Q), out, MD);
}
